// Net_25769804312
// MI455X (gfx1250) — matrix operand order repaired and checked by reading, not compiled, not run on hardware
//
#include <hip/hip_runtime.h>
#include <cstdint>
#include <cstddef>

#define BGRAPH 16
#define NNODE  2048
#define KNN    8
#define HDIM   128
#define FIN    4
#define NTOT   (BGRAPH * NNODE)   // 32768 nodes

typedef __bf16 bf16;
typedef __attribute__((ext_vector_type(16))) __bf16 v16bf;
typedef __attribute__((ext_vector_type(8)))  float  v8f;

// Trivial 16-byte chunk (avoid HIP uint4 in unions — it has user ctors).
struct alignas(16) Q16 { unsigned int u[4]; };

union Frag { v16bf v; Q16 q[2]; };   // 16 bf16 = one WMMA A/B fragment per lane
union U8   { Q16 q; bf16 h[8]; };    // 8 contiguous bf16

static __device__ __forceinline__ v8f wmma_bf16(v16bf a, v16bf b, v8f c) {
    // D = A(16x32 bf16) * B(32x16 bf16) + C(16x16 f32)
    return __builtin_amdgcn_wmma_f32_16x16x32_bf16(false, a, false, b,
                                                   (short)0, c, false, false);
}

// ---------------------------------------------------------------------------
// Weight prep: f32 [fi, fo] row-major  ->  bf16 transposed [fo, fi]
// ---------------------------------------------------------------------------
__global__ void tpose_bf16(const float* __restrict__ W, bf16* __restrict__ Wt,
                           int fi, int fo) {
    int t = blockIdx.x * blockDim.x + threadIdx.x;
    if (t >= fi * fo) return;
    int k = t / fo, n = t % fo;
    Wt[(size_t)n * fi + k] = (bf16)W[t];
}

// ---------------------------------------------------------------------------
// Encoder: h = x(4) @ W_enc(4x128) + b_enc -> bf16 features
// ---------------------------------------------------------------------------
__global__ void encoder_kernel(const float* __restrict__ x,
                               const float* __restrict__ W,
                               const float* __restrict__ b,
                               bf16* __restrict__ feat) {
    int t = blockIdx.x * blockDim.x + threadIdx.x;   // NTOT*128 threads exactly
    int node = t >> 7, c = t & 127;
    float s = b[c];
#pragma unroll
    for (int f = 0; f < FIN; ++f) s += x[node * FIN + f] * W[f * HDIM + c];
    feat[t] = (bf16)s;
}

// ---------------------------------------------------------------------------
// Per-node squared norms (for distance ranking)
// ---------------------------------------------------------------------------
__global__ void sqnorm_kernel(const bf16* __restrict__ feat, float* __restrict__ sq) {
    int n = blockIdx.x * blockDim.x + threadIdx.x;
    if (n >= NTOT) return;
    const bf16* r = feat + (size_t)n * HDIM;
    float s = 0.f;
#pragma unroll 8
    for (int f = 0; f < HDIM; ++f) { float v = (float)r[f]; s += v * v; }
    sq[n] = s;
}

// ---------------------------------------------------------------------------
// kNN: per wave, TWO 16-row tiles (32 rows) -> each B fragment feeds 2 WMMAs.
// Rank by sq[j] - 2*dot(i,j); every lane owns one row for the top-8 scan
// (register-resident sorted insert). LDS tile padded to 17 floats/row so the
// stride-17 per-lane row scan is bank-conflict-free.
// Grid: BGRAPH*8 blocks of 256 threads (8 waves x 32 rows = 256 rows/block).
// ---------------------------------------------------------------------------
__global__ void knn_kernel(const bf16* __restrict__ feat,
                           const float* __restrict__ sq,
                           int* __restrict__ idx) {
    __shared__ float dtile[8][32][17];
    int lane = threadIdx.x & 31;
    int w    = threadIdx.x >> 5;
    int g    = blockIdx.x >> 3;                 // 8 blocks per graph
    int mbase = ((blockIdx.x & 7) * 8 + w) * 32;
    int col16 = lane & 15;
    int aoff  = (lane < 16) ? 0 : 8;            // A layout: K {0..7,16..23}/{8..15,24..31}
    int koffB = (lane < 16) ? 0 : 8;            // B layout: K {0..7,16..23}/{8..15,24..31}
    int rbase = (lane < 16) ? 0 : 8;            // C layout: rows r / r+8

    // A fragments for this wave's 2 x 16 rows, K = 128 in 4 chunks of 32.
    const bf16* arow0 = feat + (size_t)(g * NNODE + mbase + col16) * HDIM;
    const bf16* arow1 = arow0 + (size_t)16 * HDIM;
    Frag a0[4], a1[4];
#pragma unroll
    for (int c = 0; c < 4; ++c) {
        a0[c].q[0] = *(const Q16*)(arow0 + c * 32 + aoff);
        a0[c].q[1] = *(const Q16*)(arow0 + c * 32 + aoff + 16);
        a1[c].q[0] = *(const Q16*)(arow1 + c * 32 + aoff);
        a1[c].q[1] = *(const Q16*)(arow1 + c * 32 + aoff + 16);
    }

    float bd[8]; int bi[8];
#pragma unroll
    for (int s = 0; s < 8; ++s) { bd[s] = 3.4e38f; bi[s] = g * NNODE; }

    for (int cb = 0; cb < NNODE / 16; ++cb) {
        const bf16* brow = feat + (size_t)(g * NNODE + cb * 16 + col16) * HDIM;
        v8f acc0 = {}, acc1 = {};
#pragma unroll
        for (int c = 0; c < 4; ++c) {
            Frag bf;
            bf.q[0] = *(const Q16*)(brow + c * 32 + koffB);
            bf.q[1] = *(const Q16*)(brow + c * 32 + koffB + 16);
            acc0 = wmma_bf16(a0[c].v, bf.v, acc0);
            acc1 = wmma_bf16(a1[c].v, bf.v, acc1);
        }
        float sqn = sq[g * NNODE + cb * 16 + col16];
#pragma unroll
        for (int r = 0; r < 8; ++r) {
            dtile[w][r + rbase][col16]      = sqn - 2.f * acc0[r];
            dtile[w][16 + r + rbase][col16] = sqn - 2.f * acc1[r];
        }
        __syncthreads();
        // Every lane owns one of the 32 rows: full-wave scan.
#pragma unroll
        for (int col = 0; col < 16; ++col) {
            float d = dtile[w][lane][col];
            if (d < bd[7]) {                       // sorted insert (unrolled -> VGPRs)
                bd[7] = d; bi[7] = g * NNODE + cb * 16 + col;
#pragma unroll
                for (int p = 7; p >= 1; --p) {
                    if (bd[p] < bd[p - 1]) {
                        float td = bd[p]; bd[p] = bd[p - 1]; bd[p - 1] = td;
                        int   ti = bi[p]; bi[p] = bi[p - 1]; bi[p - 1] = ti;
                    }
                }
            }
        }
        __syncthreads();
    }
    int node = g * NNODE + mbase + lane;
#pragma unroll
    for (int s = 0; s < 8; ++s) idx[(size_t)node * KNN + s] = bi[s];
}

// ---------------------------------------------------------------------------
// EdgeConv: wave = 2 nodes = 16 edges (M=16). e=[x_i, x_j-x_i] (K=256),
// GEMM1 -> ReLU -> GEMM2 -> ReLU -> per-lane max over the 8 rows of each node.
// FINAL: second layer is a 128->1 dot + shfl max + sigmoid into d_out.
// Grid: NTOT/16 blocks of 256 threads.
// ---------------------------------------------------------------------------
template <bool FINAL>
__global__ void edgeconv_kernel(const bf16* __restrict__ feat,
                                const int*  __restrict__ idx,
                                const bf16* __restrict__ WaT,  // [128][256] bf16
                                const float* __restrict__ ba,
                                const bf16* __restrict__ WbT,  // [128][128] or [128] bf16
                                const float* __restrict__ bb,
                                bf16* __restrict__ outFeat,
                                float* __restrict__ outFinal) {
    __shared__ __align__(16) bf16 smem[8][2304];   // per wave: 2x128 self + 16x128 diff/h1
    int lane = threadIdx.x & 31;
    int w    = threadIdx.x >> 5;
    int nbase = blockIdx.x * 16 + w * 2;
    bf16* selfp = &smem[w][0];
    bf16* workp = &smem[w][256];

    // Stage self rows (2 x 128 bf16): 8 bf16 per lane.
    {
        int row = lane >> 4;
        int f0  = (lane & 15) * 8;
        *(Q16*)(selfp + row * HDIM + f0) =
            *(const Q16*)(feat + (size_t)(nbase + row) * HDIM + f0);
    }
    // Stage diffs x_j - x_i (16 x 128 bf16): 2 lanes per edge, 64 feats each.
    {
        int e = lane >> 1, hf = lane & 1;
        int inode = nbase + (e >> 3);
        int j = idx[(size_t)inode * KNN + (e & 7)];
        const bf16* rj = feat + (size_t)j * HDIM + hf * 64;
        const bf16* ri = feat + (size_t)inode * HDIM + hf * 64;
        bf16* rd = workp + e * HDIM + hf * 64;
#pragma unroll
        for (int f0 = 0; f0 < 64; f0 += 8) {
            U8 aa, bb2, rr;
            aa.q  = *(const Q16*)(rj + f0);
            bb2.q = *(const Q16*)(ri + f0);
#pragma unroll
            for (int t = 0; t < 8; ++t)
                rr.h[t] = (bf16)((float)aa.h[t] - (float)bb2.h[t]);
            *(Q16*)(rd + f0) = rr.q;
        }
    }
    __syncthreads();

    int m     = lane & 15;                 // edge (row) index within tile
    int aoff  = (lane < 16) ? 0 : 8;
    int koffB = (lane < 16) ? 0 : 8;
    int col16 = lane & 15;
    int rbase = (lane < 16) ? 0 : 8;

    // A fragments for e = [self | diff], K = 256 in 8 chunks.
    Frag a1[8];
#pragma unroll
    for (int c = 0; c < 8; ++c) {
        const bf16* src = (c < 4) ? (selfp + (m >> 3) * HDIM + c * 32)
                                  : (workp + m * HDIM + (c - 4) * 32);
        a1[c].q[0] = *(const Q16*)(src + aoff);
        a1[c].q[1] = *(const Q16*)(src + aoff + 16);
    }
    __syncthreads();

    // GEMM1: (16x256) @ (256x128), bias + ReLU, h1 -> LDS (overwrites diffs).
#pragma unroll 1
    for (int nt = 0; nt < 8; ++nt) {
        int col = nt * 16 + col16;
        v8f acc = {};
#pragma unroll
        for (int c = 0; c < 8; ++c) {
            Frag bf;
            const bf16* src = WaT + (size_t)col * 256 + c * 32 + koffB;
            bf.q[0] = *(const Q16*)src;
            bf.q[1] = *(const Q16*)(src + 16);
            acc = wmma_bf16(a1[c].v, bf.v, acc);
        }
        float bias = ba[col];
#pragma unroll
        for (int r = 0; r < 8; ++r) {
            float v = acc[r] + bias;
            v = v > 0.f ? v : 0.f;
            workp[(r + rbase) * HDIM + col] = (bf16)v;
        }
    }
    __syncthreads();

    if (!FINAL) {
        // GEMM2: (16x128) @ (128x128), bias + ReLU, max over 8 rows per node.
        Frag a2[4];
#pragma unroll
        for (int c = 0; c < 4; ++c) {
            const bf16* src = workp + m * HDIM + c * 32;
            a2[c].q[0] = *(const Q16*)(src + aoff);
            a2[c].q[1] = *(const Q16*)(src + aoff + 16);
        }
        int node = nbase + ((lane < 16) ? 0 : 1);
#pragma unroll 1
        for (int nt = 0; nt < 8; ++nt) {
            int col = nt * 16 + col16;
            v8f acc = {};
#pragma unroll
            for (int c = 0; c < 4; ++c) {
                Frag bf;
                const bf16* src = WbT + (size_t)col * HDIM + c * 32 + koffB;
                bf.q[0] = *(const Q16*)src;
                bf.q[1] = *(const Q16*)(src + 16);
                acc = wmma_bf16(a2[c].v, bf.v, acc);
            }
            float bias = bb[col];
            float best = 0.f;                       // values are ReLU'd (>= 0)
#pragma unroll
            for (int r = 0; r < 8; ++r) {
                float v = acc[r] + bias;
                v = v > 0.f ? v : 0.f;
                best = v > best ? v : best;
            }
            outFeat[(size_t)node * HDIM + col] = (bf16)best;
        }
    } else {
        // 128 -> 1 dot per edge, ReLU, max over 8 edges, sigmoid.
        float v = -3.4e38f;
        if (lane < 16) {
            float dot = bb[0];
            const bf16* hr = workp + lane * HDIM;
#pragma unroll 8
            for (int f = 0; f < HDIM; ++f)
                dot += (float)hr[f] * (float)WbT[f];
            v = dot > 0.f ? dot : 0.f;
        }
        v = fmaxf(v, __shfl_xor(v, 1, 32));
        v = fmaxf(v, __shfl_xor(v, 2, 32));
        v = fmaxf(v, __shfl_xor(v, 4, 32));
        if (lane == 0) outFinal[nbase]     = 1.f / (1.f + __expf(-v));
        if (lane == 8) outFinal[nbase + 1] = 1.f / (1.f + __expf(-v));
    }
}

// ---------------------------------------------------------------------------
extern "C" void kernel_launch(void* const* d_in, const int* in_sizes, int n_in,
                              void* d_out, int out_size, void* d_ws, size_t ws_size,
                              hipStream_t stream) {
    (void)in_sizes; (void)n_in; (void)out_size; (void)ws_size;
    const float* x     = (const float*)d_in[0];
    const float* W_enc = (const float*)d_in[3];
    const float* b_enc = (const float*)d_in[4];
    const float* W1a = (const float*)d_in[5];  const float* b1a = (const float*)d_in[6];
    const float* W1b = (const float*)d_in[7];  const float* b1b = (const float*)d_in[8];
    const float* W2a = (const float*)d_in[9];  const float* b2a = (const float*)d_in[10];
    const float* W2b = (const float*)d_in[11]; const float* b2b = (const float*)d_in[12];
    const float* W5a = (const float*)d_in[13]; const float* b5a = (const float*)d_in[14];
    const float* W5b = (const float*)d_in[15]; const float* b5b = (const float*)d_in[16];

    char* ws = (char*)d_ws;
    size_t off = 0;
    auto alloc = [&](size_t bytes) -> char* {
        char* p = ws + off;
        off += (bytes + 255) & ~(size_t)255;
        return p;
    };
    bf16* featA = (bf16*)alloc((size_t)NTOT * HDIM * sizeof(bf16));
    bf16* featB = (bf16*)alloc((size_t)NTOT * HDIM * sizeof(bf16));
    float* sq   = (float*)alloc((size_t)NTOT * sizeof(float));
    int*  idxb  = (int*)alloc((size_t)NTOT * KNN * sizeof(int));
    bf16* W1aT  = (bf16*)alloc((size_t)256 * 128 * sizeof(bf16));
    bf16* W1bT  = (bf16*)alloc((size_t)128 * 128 * sizeof(bf16));
    bf16* W2aT  = (bf16*)alloc((size_t)256 * 128 * sizeof(bf16));
    bf16* W2bT  = (bf16*)alloc((size_t)128 * 128 * sizeof(bf16));
    bf16* W5aT  = (bf16*)alloc((size_t)256 * 128 * sizeof(bf16));
    bf16* w5bT  = (bf16*)alloc((size_t)128 * sizeof(bf16));

    // Weight prep (bf16, transposed to [fo][fi]).
    tpose_bf16<<<(256 * 128 + 255) / 256, 256, 0, stream>>>(W1a, W1aT, 256, 128);
    tpose_bf16<<<(128 * 128 + 255) / 256, 256, 0, stream>>>(W1b, W1bT, 128, 128);
    tpose_bf16<<<(256 * 128 + 255) / 256, 256, 0, stream>>>(W2a, W2aT, 256, 128);
    tpose_bf16<<<(128 * 128 + 255) / 256, 256, 0, stream>>>(W2b, W2bT, 128, 128);
    tpose_bf16<<<(256 * 128 + 255) / 256, 256, 0, stream>>>(W5a, W5aT, 256, 128);
    tpose_bf16<<<1, 256, 0, stream>>>(W5b, w5bT, 128, 1);

    // Encoder.
    encoder_kernel<<<(NTOT * HDIM) / 256, 256, 0, stream>>>(x, W_enc, b_enc, featA);

    // Conv 1: featA -> featB
    sqnorm_kernel<<<NTOT / 256, 256, 0, stream>>>(featA, sq);
    knn_kernel<<<BGRAPH * 8, 256, 0, stream>>>(featA, sq, idxb);
    edgeconv_kernel<false><<<NTOT / 16, 256, 0, stream>>>(featA, idxb, W1aT, b1a,
                                                          W1bT, b1b, featB, nullptr);
    // Conv 2: featB -> featA
    sqnorm_kernel<<<NTOT / 256, 256, 0, stream>>>(featB, sq);
    knn_kernel<<<BGRAPH * 8, 256, 0, stream>>>(featB, sq, idxb);
    edgeconv_kernel<false><<<NTOT / 16, 256, 0, stream>>>(featB, idxb, W2aT, b2a,
                                                          W2bT, b2b, featA, nullptr);
    // Conv 3: featA -> d_out (sigmoid scores)
    sqnorm_kernel<<<NTOT / 256, 256, 0, stream>>>(featA, sq);
    knn_kernel<<<BGRAPH * 8, 256, 0, stream>>>(featA, sq, idxb);
    edgeconv_kernel<true><<<NTOT / 16, 256, 0, stream>>>(featA, idxb, W5aT, b5a,
                                                         w5bT, b5b, nullptr,
                                                         (float*)d_out);
}
